// RelativeTransformerDecoderLayer_54107997995225
// MI455X (gfx1250) — matrix operand order repaired and checked by reading, not compiled, not run on hardware
//
#include <hip/hip_runtime.h>
#include <math.h>

// ---------------------------------------------------------------------------
// RelativeTransformerDecoderLayer for MI455X (gfx1250, wave32, WMMA).
// GEMMs: v_wmma_f32_16x16x32_bf16, ping-pong LDS tiles staged with
// GLOBAL_LOAD_ASYNC_TO_LDS_B128 (ASYNCcnt) when the toolchain exposes it.
// ---------------------------------------------------------------------------

typedef __bf16 bf16;
typedef __attribute__((ext_vector_type(16))) bf16 v16bf;
typedef __attribute__((ext_vector_type(8)))  bf16 v8bf;
typedef __attribute__((ext_vector_type(8)))  float v8f;
typedef __attribute__((ext_vector_type(4)))  int   v4i;

#define AS1 __attribute__((address_space(1)))
#define AS3 __attribute__((address_space(3)))

#define T_SEQ   2048
#define D_MODEL 1024
#define NHEAD   16
#define DHEAD   64
#define DFF     4096

#if defined(__gfx1250__) && \
    __has_builtin(__builtin_amdgcn_global_load_async_to_lds_b128) && \
    __has_builtin(__builtin_amdgcn_s_wait_asynccnt)
#define USE_ASYNC_LDS 1
#else
#define USE_ASYNC_LDS 0
#endif

__device__ __forceinline__ void async_b128(const bf16* gsrc, bf16* ldst) {
#if USE_ASYNC_LDS
    __builtin_amdgcn_global_load_async_to_lds_b128(
        (AS1 v4i*)gsrc, (AS3 v4i*)ldst, 0, 0);
#else
    *(v8bf*)ldst = *(const v8bf*)gsrc;     // fallback: VGPR roundtrip
#endif
}
__device__ __forceinline__ void wait_async_lds() {
#if USE_ASYNC_LDS
    __builtin_amdgcn_s_wait_asynccnt(0);
#endif
}

// ---------------------------------------------------------------------------
// bf16 WMMA GEMM:  C[M,N] (+bias)(+relu)(+resid) = A[M,K] * B[K,N]
//  - A (AF32==0): bf16, k-contiguous rows, row stride ars      -> async stage
//  - A (AF32==1): fp32, arbitrary element strides (probs x V)  -> scalar stage
//  - B: always N-major: element (k,n) at B[n*bcs + k]          -> async stage
// Block tile 128x128, 8 wave32 waves, each wave 64x32 = 4x2 WMMA tiles.
// Double-buffered LDS: prefetch tile i+1 while computing tile i.
// ---------------------------------------------------------------------------
template <int AF32>
__global__ __launch_bounds__(256) void gemm_wmma_k(
    const bf16*  __restrict__ A,   long ars,
    const float* __restrict__ A32, long a32rs, long a32cs,
    const bf16*  __restrict__ B,   long bcs,
    float* __restrict__ C, long crs, long ccs,
    const float* __restrict__ bias,
    const float* __restrict__ resid, long rrs,
    int relu, int M, int N, int K)
{
    // padded rows: 40 elems = 80 B (multiple of 16 B -> aligned v8bf loads)
    __shared__ __align__(16) bf16 As[2][128][40];
    __shared__ __align__(16) bf16 Bs[2][128][40];   // transposed: Bs[n][k]

    const int  tid  = threadIdx.x;
    const long row0 = (long)blockIdx.y * 128;
    const long col0 = (long)blockIdx.x * 128;
    const int  lane = tid & 31;
    const int  wave = tid >> 5;
    const int  half = lane >> 4;
    const int  r    = lane & 15;
    const int  wm   = (wave & 1) * 64;
    const int  wn   = (wave >> 1) * 32;

    v8f acc[4][2];
    for (int mt = 0; mt < 4; ++mt)
        for (int nt = 0; nt < 2; ++nt)
            for (int i = 0; i < 8; ++i) acc[mt][nt][i] = 0.0f;

    auto stage = [&](int k0, int buf) {
        // B tile: 128 n-rows x 32 k  = 512 chunks of 8 bf16 (16 B)
        for (int c = tid; c < 512; c += 256) {
            int  nn = c >> 2, kc = (c & 3) * 8;
            long gn = col0 + nn; if (gn >= N) gn = N - 1;   // safe clamp
            async_b128(&B[gn * bcs + k0 + kc], &Bs[buf][nn][kc]);
        }
        if constexpr (AF32) {
            for (int idx = tid; idx < 128 * 32; idx += 256) {
                int  rr = idx >> 5, cc = idx & 31;
                long gr = row0 + rr, gk = (long)k0 + cc;
                float v = 0.0f;
                if (gr < M && gk < K) v = A32[gr * a32rs + gk * a32cs];
                As[buf][rr][cc] = (bf16)v;
            }
        } else {
            for (int c = tid; c < 512; c += 256) {
                int  rr = c >> 2, kc = (c & 3) * 8;
                long gr = row0 + rr; if (gr >= M) gr = M - 1;
                async_b128(&A[gr * ars + k0 + kc], &As[buf][rr][kc]);
            }
        }
    };

    auto compute = [&](int buf) {
        v16bf af[4], bfr[2];
        for (int mt = 0; mt < 4; ++mt) {
            // A lane layout: row=r, K={8h..8h+7, 8h+16..8h+23}
            const v8bf* p = (const v8bf*)&As[buf][wm + mt * 16 + r][half * 8];
            af[mt] = __builtin_shufflevector(p[0], p[2],
                     0,1,2,3,4,5,6,7,8,9,10,11,12,13,14,15);
        }
        for (int nt = 0; nt < 2; ++nt) {
            // B lane layout: col=r, K={8h..8h+7, 8h+16..8h+23}
            const v8bf* p = (const v8bf*)&Bs[buf][wn + nt * 16 + r][half * 8];
            bfr[nt] = __builtin_shufflevector(p[0], p[2],
                      0,1,2,3,4,5,6,7,8,9,10,11,12,13,14,15);
        }
        for (int mt = 0; mt < 4; ++mt)
            for (int nt = 0; nt < 2; ++nt)
                acc[mt][nt] = __builtin_amdgcn_wmma_f32_16x16x32_bf16(
                    false, af[mt], false, bfr[nt],
                    (short)0, acc[mt][nt], false, false);
    };

    const int NK = (K + 31) >> 5;
    stage(0, 0);
    wait_async_lds();
    __syncthreads();
    for (int i = 0; i < NK; ++i) {
        const int cur = i & 1;
        if (i + 1 < NK) stage((i + 1) * 32, cur ^ 1);   // prefetch next tile
        compute(cur);                                    // overlap with async
        wait_async_lds();
        __syncthreads();
    }

    // epilogue: C lane layout: N = r, VGPR vi -> M = vi + 8*half
    for (int nt = 0; nt < 2; ++nt) {
        long gc = col0 + wn + nt * 16 + r;
        if (gc >= N) continue;
        float bv = bias ? bias[gc] : 0.0f;
        for (int mt = 0; mt < 4; ++mt) {
            for (int vi = 0; vi < 8; ++vi) {
                long gr = row0 + wm + mt * 16 + vi + half * 8;
                if (gr >= M) continue;
                float v = acc[mt][nt][vi] + bv;
                if (relu)  v = fmaxf(v, 0.0f);
                if (resid) v += resid[gr * rrs + gc];
                C[gr * crs + gc * ccs] = v;
            }
        }
    }
}

// ---------------------------------------------------------------------------
// LayerNorm (fp32 in) -> bf16 out. One block per row, D=1024.
// ---------------------------------------------------------------------------
__global__ __launch_bounds__(256) void layernorm_to_bf16(
    const float* __restrict__ x, const float* __restrict__ g,
    const float* __restrict__ b, bf16* __restrict__ out)
{
    __shared__ float red[256];
    const int row = blockIdx.x, tid = threadIdx.x;
    float v[4];
    float s = 0.0f;
    for (int t = 0; t < 4; ++t) {
        v[t] = x[(long)row * D_MODEL + tid + t * 256];
        s += v[t];
    }
    red[tid] = s; __syncthreads();
    for (int st = 128; st > 0; st >>= 1) {
        if (tid < st) red[tid] += red[tid + st];
        __syncthreads();
    }
    float mean = red[0] * (1.0f / D_MODEL);
    __syncthreads();
    s = 0.0f;
    for (int t = 0; t < 4; ++t) { float d = v[t] - mean; s += d * d; }
    red[tid] = s; __syncthreads();
    for (int st = 128; st > 0; st >>= 1) {
        if (tid < st) red[tid] += red[tid + st];
        __syncthreads();
    }
    float inv = rsqrtf(red[0] * (1.0f / D_MODEL) + 1e-5f);
    for (int t = 0; t < 4; ++t) {
        int c = tid + t * 256;
        out[(long)row * D_MODEL + c] = (bf16)((v[t] - mean) * inv * g[c] + b[c]);
    }
}

// ---------------------------------------------------------------------------
// Fused mask + rel_shift gather + softmax for one head.
// AC in ac[i*T + j] (contiguous), BD_raw in bdr[i*T + c].
// rel_shift: BD[i,j] = BD_raw[i, T-1-(i-j)] for j<=i; future -> prob 0.
// Writes normalized probs into the interleaved (T,T,B,H) output layout.
// ---------------------------------------------------------------------------
__global__ __launch_bounds__(256) void attn_softmax(
    float* __restrict__ probs, const float* __restrict__ ac,
    const float* __restrict__ bdr, int h)
{
    __shared__ float red[256];
    const int i = blockIdx.x, tid = threadIdx.x;
    float vals[8];
    float mx = -__builtin_inff();
    for (int t = 0; t < 8; ++t) {
        int j = tid + t * 256;
        float s = -__builtin_inff();
        if (j <= i) {
            float a = ac[(long)i * T_SEQ + j];
            float bd = bdr[(long)i * T_SEQ + (T_SEQ - 1 - i + j)];
            s = (a + bd) * 0.125f;             // 1/sqrt(64)
        }
        vals[t] = s;
        mx = fmaxf(mx, s);
    }
    red[tid] = mx; __syncthreads();
    for (int st = 128; st > 0; st >>= 1) {
        if (tid < st) red[tid] = fmaxf(red[tid], red[tid + st]);
        __syncthreads();
    }
    mx = red[0]; __syncthreads();
    float sum = 0.0f;
    for (int t = 0; t < 8; ++t) {
        float e = __expf(vals[t] - mx);        // exp(-inf)=0 for masked
        vals[t] = e;
        sum += e;
    }
    red[tid] = sum; __syncthreads();
    for (int st = 128; st > 0; st >>= 1) {
        if (tid < st) red[tid] += red[tid + st];
        __syncthreads();
    }
    float inv = 1.0f / red[0];
    for (int t = 0; t < 8; ++t) {
        int j = tid + t * 256;
        probs[((long)i * T_SEQ + j) * NHEAD + h] = vals[t] * inv;
    }
}

// ---------------------------------------------------------------------------
// Split QKV heads, add r_w/r_r biases, convert to per-head bf16 tensors.
// V is written transposed (Vt[h*64+d][t]) so probs x V streams k-contiguous.
// ---------------------------------------------------------------------------
__global__ __launch_bounds__(256) void prep_heads(
    const float* __restrict__ wh, const float* __restrict__ rk,
    const float* __restrict__ rwb, const float* __restrict__ rrb,
    bf16* __restrict__ Qw, bf16* __restrict__ Qr, bf16* __restrict__ Kb,
    bf16* __restrict__ Vt, bf16* __restrict__ RKb)
{
    long idx = (long)blockIdx.x * 256 + threadIdx.x;   // t*1024 + hd
    int  t  = (int)(idx >> 10);
    int  hd = (int)(idx & 1023);
    int  h  = hd >> 6;
    float q = wh[(long)t * 3072 + hd];
    float k = wh[(long)t * 3072 + 1024 + hd];
    float v = wh[(long)t * 3072 + 2048 + hd];
    long o  = (((long)h * T_SEQ + t) << 6) + (hd & 63);
    Qw[o]  = (bf16)(q + rwb[hd]);
    Qr[o]  = (bf16)(q + rrb[hd]);
    Kb[o]  = (bf16)k;
    Vt[(long)hd * T_SEQ + t] = (bf16)v;                // transposed layout
    RKb[o] = (bf16)rk[idx];
}

__global__ void f32_to_bf16_k(const float* __restrict__ in,
                              bf16* __restrict__ out, long n)
{
    long i = (long)blockIdx.x * 256 + threadIdx.x;
    if (i < n) out[i] = (bf16)in[i];
}

// fp32 [K,N] -> bf16 transposed [N,K] (weights become N-major for the GEMM)
__global__ void cvt_transpose_k(const float* __restrict__ in,
                                bf16* __restrict__ out, int K, int N)
{
    long idx = (long)blockIdx.x * 256 + threadIdx.x;   // over K*N
    long k = idx / N, n = idx % N;
    out[n * (long)K + k] = (bf16)in[idx];
}

// ---------------------------------------------------------------------------
// Host side
// ---------------------------------------------------------------------------
static inline void launch_gemm(hipStream_t s,
    const bf16* A, long ars,
    const float* A32, long a32rs, long a32cs,
    const bf16* B, long bcs,
    float* C, long crs, long ccs,
    const float* bias, const float* resid, long rrs, int relu,
    int M, int N, int K)
{
    dim3 grid((N + 127) / 128, (M + 127) / 128);
    if (A32)
        gemm_wmma_k<1><<<grid, 256, 0, s>>>(A, ars, A32, a32rs, a32cs, B, bcs,
                                            C, crs, ccs, bias, resid, rrs,
                                            relu, M, N, K);
    else
        gemm_wmma_k<0><<<grid, 256, 0, s>>>(A, ars, A32, a32rs, a32cs, B, bcs,
                                            C, crs, ccs, bias, resid, rrs,
                                            relu, M, N, K);
}

extern "C" void kernel_launch(void* const* d_in, const int* in_sizes, int n_in,
                              void* d_out, int out_size, void* d_ws, size_t ws_size,
                              hipStream_t stream) {
    (void)in_sizes; (void)n_in; (void)out_size; (void)ws_size;
    const float* input    = (const float*)d_in[0];
    const float* pos      = (const float*)d_in[1];
    const float* r_w_bias = (const float*)d_in[2];
    const float* r_r_bias = (const float*)d_in[3];
    /* d_in[4] = mask: causal, hardcoded in attn_softmax */
    const float* ln1_g = (const float*)d_in[5];
    const float* ln1_b = (const float*)d_in[6];
    const float* qkv_w = (const float*)d_in[7];
    const float* r_w   = (const float*)d_in[8];
    const float* o_w   = (const float*)d_in[9];
    const float* ln2_g = (const float*)d_in[10];
    const float* ln2_b = (const float*)d_in[11];
    const float* ff_w1 = (const float*)d_in[12];
    const float* ff_b1 = (const float*)d_in[13];
    const float* ff_w2 = (const float*)d_in[14];
    const float* ff_b2 = (const float*)d_in[15];

    float* x_out = (float*)d_out;                                  // (T,B,D)
    float* probs = (float*)d_out + (size_t)T_SEQ * D_MODEL;        // (T,T,B,H)

    char*  ws  = (char*)d_ws;
    size_t off = 0;
    auto alloc = [&](size_t bytes) -> void* {
        size_t o = (off + 255) & ~(size_t)255;
        off = o + bytes;
        return (void*)(ws + o);
    };

    // transposed bf16 weights (N-major)
    bf16*  qkvw_t = (bf16*)alloc((size_t)D_MODEL * 3072 * 2);
    bf16*  rw_t   = (bf16*)alloc((size_t)D_MODEL * D_MODEL * 2);
    bf16*  ow_t   = (bf16*)alloc((size_t)D_MODEL * D_MODEL * 2);
    bf16*  ffw1_t = (bf16*)alloc((size_t)D_MODEL * DFF * 2);
    bf16*  ffw2_t = (bf16*)alloc((size_t)DFF * D_MODEL * 2);
    bf16*  pos_b  = (bf16*)alloc((size_t)T_SEQ * D_MODEL * 2);
    bf16*  ln1_o  = (bf16*)alloc((size_t)T_SEQ * D_MODEL * 2);
    float* wheads = (float*)alloc((size_t)T_SEQ * 3072 * 4);   // reused: ff1_f
    float* rk_f   = (float*)alloc((size_t)T_SEQ * D_MODEL * 4);
    bf16*  Qw     = (bf16*)alloc((size_t)NHEAD * T_SEQ * DHEAD * 2);
    bf16*  Qr     = (bf16*)alloc((size_t)NHEAD * T_SEQ * DHEAD * 2);
    bf16*  Kb     = (bf16*)alloc((size_t)NHEAD * T_SEQ * DHEAD * 2);
    bf16*  Vt     = (bf16*)alloc((size_t)NHEAD * T_SEQ * DHEAD * 2);
    bf16*  RKb    = (bf16*)alloc((size_t)NHEAD * T_SEQ * DHEAD * 2);
    float* ac_f   = (float*)alloc((size_t)T_SEQ * T_SEQ * 4);
    float* bdr    = (float*)alloc((size_t)T_SEQ * T_SEQ * 4);  // reused: ff1_b
    float* attnv  = (float*)alloc((size_t)T_SEQ * D_MODEL * 4);
    bf16*  attnv_b= (bf16*)alloc((size_t)T_SEQ * D_MODEL * 2);
    float* x_f    = (float*)alloc((size_t)T_SEQ * D_MODEL * 4);
    bf16*  y_b    = (bf16*)alloc((size_t)T_SEQ * D_MODEL * 2);
    // lifetimes disjoint: wheads+rk_f (32MB) -> ff1 fp32; bdr (16MB) -> ff1 bf16
    float* ff1_f  = wheads;
    bf16*  ff1_b  = (bf16*)bdr;

    // ---- weight conversions: fp32 [K,N] -> bf16 [N,K] ----
    cvt_transpose_k<<<(D_MODEL * 3072) / 256, 256, 0, stream>>>(qkv_w, qkvw_t, D_MODEL, 3072);
    cvt_transpose_k<<<(D_MODEL * D_MODEL) / 256, 256, 0, stream>>>(r_w, rw_t, D_MODEL, D_MODEL);
    cvt_transpose_k<<<(D_MODEL * D_MODEL) / 256, 256, 0, stream>>>(o_w, ow_t, D_MODEL, D_MODEL);
    cvt_transpose_k<<<(D_MODEL * DFF) / 256, 256, 0, stream>>>(ff_w1, ffw1_t, D_MODEL, DFF);
    cvt_transpose_k<<<(DFF * D_MODEL) / 256, 256, 0, stream>>>(ff_w2, ffw2_t, DFF, D_MODEL);
    f32_to_bf16_k<<<(T_SEQ * D_MODEL) / 256, 256, 0, stream>>>(pos, pos_b, (long)T_SEQ * D_MODEL);

    // ---- LN1, QKV projection, r_k projection ----
    layernorm_to_bf16<<<T_SEQ, 256, 0, stream>>>(input, ln1_g, ln1_b, ln1_o);
    launch_gemm(stream, ln1_o, D_MODEL, nullptr, 0, 0, qkvw_t, D_MODEL,
                wheads, 3072, 1, nullptr, nullptr, 0, 0, T_SEQ, 3072, D_MODEL);
    launch_gemm(stream, pos_b, D_MODEL, nullptr, 0, 0, rw_t, D_MODEL,
                rk_f, D_MODEL, 1, nullptr, nullptr, 0, 0, T_SEQ, D_MODEL, D_MODEL);
    prep_heads<<<(T_SEQ * D_MODEL) / 256, 256, 0, stream>>>(
        wheads, rk_f, r_w_bias, r_r_bias, Qw, Qr, Kb, Vt, RKb);

    // ---- attention per head: AC, BD, softmax, probs x V ----
    for (int h = 0; h < NHEAD; ++h) {
        const bf16* Qwh = Qw  + (size_t)h * T_SEQ * DHEAD;
        const bf16* Qrh = Qr  + (size_t)h * T_SEQ * DHEAD;
        const bf16* Kbh = Kb  + (size_t)h * T_SEQ * DHEAD;
        const bf16* Vth = Vt  + (size_t)h * DHEAD * T_SEQ;
        const bf16* RKh = RKb + (size_t)h * T_SEQ * DHEAD;
        // AC = (Q + r_w_bias) x K^T  (K is N-major: bcs=DHEAD)
        launch_gemm(stream, Qwh, DHEAD, nullptr, 0, 0, Kbh, DHEAD,
                    ac_f, T_SEQ, 1, nullptr, nullptr, 0, 0, T_SEQ, T_SEQ, DHEAD);
        // BD_raw = (Q + r_r_bias) x r_k^T
        launch_gemm(stream, Qrh, DHEAD, nullptr, 0, 0, RKh, DHEAD,
                    bdr, T_SEQ, 1, nullptr, nullptr, 0, 0, T_SEQ, T_SEQ, DHEAD);
        attn_softmax<<<T_SEQ, 256, 0, stream>>>(probs, ac_f, bdr, h);
        // attn_vec_h = probs_h x V_h  (fp32 interleaved A; V transposed bf16)
        launch_gemm(stream, nullptr, 0, probs + h, (long)T_SEQ * NHEAD, NHEAD,
                    Vth, T_SEQ, attnv + h * DHEAD, D_MODEL, 1,
                    nullptr, nullptr, 0, 0, T_SEQ, DHEAD, T_SEQ);
    }

    // ---- output projection + residual ----
    f32_to_bf16_k<<<(T_SEQ * D_MODEL) / 256, 256, 0, stream>>>(attnv, attnv_b, (long)T_SEQ * D_MODEL);
    launch_gemm(stream, attnv_b, D_MODEL, nullptr, 0, 0, ow_t, D_MODEL,
                x_f, D_MODEL, 1, nullptr, input, D_MODEL, 0,
                T_SEQ, D_MODEL, D_MODEL);

    // ---- FFN block ----
    layernorm_to_bf16<<<T_SEQ, 256, 0, stream>>>(x_f, ln2_g, ln2_b, y_b);
    launch_gemm(stream, y_b, D_MODEL, nullptr, 0, 0, ffw1_t, D_MODEL,
                ff1_f, DFF, 1, ff_b1, nullptr, 0, 1, T_SEQ, DFF, D_MODEL);
    f32_to_bf16_k<<<(T_SEQ * DFF) / 256, 256, 0, stream>>>(ff1_f, ff1_b, (long)T_SEQ * DFF);
    launch_gemm(stream, ff1_b, DFF, nullptr, 0, 0, ffw2_t, DFF,
                x_out, D_MODEL, 1, ff_b2, x_f, D_MODEL, 0,
                T_SEQ, D_MODEL, DFF);
}
